// MaskEncoderHier_78898549227872
// MI455X (gfx1250) — matrix operand order repaired and checked by reading, not compiled, not run on hardware
//
#include <hip/hip_runtime.h>
#include <hip/hip_bf16.h>

typedef __attribute__((ext_vector_type(16))) _Float16 v16h;
typedef __attribute__((ext_vector_type(8)))  float    v8f;

#define B_    4
#define N_    65536
#define M1_   4096
#define K1_   32
#define M2_   1024
#define K2_   16
#define EMB_  256

// -------------------- helpers --------------------

__device__ __forceinline__ float gelu_f(float x) {
  // jax.nn.gelu approximate=True (tanh form)
  const float c0 = 0.7978845608028654f;
  return 0.5f * x * (1.0f + tanhf(c0 * (x + 0.044715f * x * x * x)));
}

// A fragment: 16x32 f16, rows r0..r0+15, K k0..k0+31, per ISA 16-bit A layout.
__device__ __forceinline__ v16h load_A_frag(const _Float16* A, int lda, int r0, int k0, int lane) {
  const int m  = lane & 15;
  const int kh = (lane >> 4) & 1;              // 0: K 0-7 & 16-23 ; 1: K 8-15 & 24-31
  const _Float16* row = A + (r0 + m) * lda + k0;
  v16h a;
#pragma unroll
  for (int i = 0; i < 8; ++i) a[i]     = row[kh * 8 + i];
#pragma unroll
  for (int i = 0; i < 8; ++i) a[8 + i] = row[16 + kh * 8 + i];
  return a;
}

// B fragment: 32x16 f16 from transposed weights WT[cout][ldw]; lane holds one column,
// K order matches A (lanes 0-15: K 0-7 & 16-23, lanes 16-31: K 8-15 & 24-31).
__device__ __forceinline__ v16h load_B_frag(const _Float16* WT, int ldw, int n0, int k0, int lane) {
  const int n  = lane & 15;
  const int kh = (lane >> 4) & 1;
  const _Float16* col = WT + (n0 + n) * ldw + k0;
  v16h b;
#pragma unroll
  for (int i = 0; i < 8; ++i) b[i]     = col[kh * 8 + i];
#pragma unroll
  for (int i = 0; i < 8; ++i) b[8 + i] = col[16 + kh * 8 + i];
  return b;
}

// D fragment: VGPR j -> row j (lanes 0-15) / row 8+j (lanes 16-31), col = lane%16.
__device__ __forceinline__ void store_D_frag(float* C, int ldc, int r0, int n0, int lane, v8f d) {
  const int n  = lane & 15;
  const int mb = (lane >> 4) ? 8 : 0;
#pragma unroll
  for (int j = 0; j < 8; ++j) C[(r0 + mb + j) * ldc + n0 + n] = d[j];
}

// C[Mrows x Ncols] = A[Mrows x Kdim] * WT^T. 4 waves; each wave owns one M-tile
// and up to 4 N-tiles; A fragment loaded once per k-step and reused across N-tiles,
// accumulators held in registers across the whole K loop.
__device__ __forceinline__ void mm_lds(const _Float16* A, int lda,
                                       const _Float16* WT, int ldw,
                                       float* C, int ldc,
                                       int Mrows, int Ncols, int Kdim,
                                       int wave, int lane) {
  const int mt = Mrows >> 4, nt = Ncols >> 4, kt = Kdim >> 5;
  const int wpm = 4 / mt;                          // waves sharing one M-tile (2 or 4)
  const int mi  = (mt == 2) ? (wave >> 1) : 0;
  const int nw  = (mt == 2) ? (wave & 1) : wave;   // wave index within M-tile group
  const int nni = nt / wpm;                        // N-tiles per wave (<= 4)

  v8f acc[4] = {};
  for (int k = 0; k < kt; ++k) {
    v16h a = load_A_frag(A, lda, mi * 16, k * 32, lane);
#pragma unroll 4
    for (int j = 0; j < nni; ++j) {
      v16h b = load_B_frag(WT, ldw, (nw + j * wpm) * 16, k * 32, lane);
      acc[j] = __builtin_amdgcn_wmma_f32_16x16x32_f16(false, a, false, b, (short)0,
                                                      acc[j], false, false);
    }
  }
#pragma unroll 4
  for (int j = 0; j < nni; ++j)
    store_D_frag(C, ldc, mi * 16, (nw + j * wpm) * 16, lane, acc[j]);
}

// x = gelu(LN(C + bias) * g + be); A[:, 0:H] = f16(x); A[:, H:2H] = colmax over rows.
// nthreads/R lanes cooperate on each row; partials reduced with lane shuffles.
__device__ __forceinline__ void ln_gelu_concat(const float* C, int ldc, const float* bias,
                                               const float* g, const float* be,
                                               _Float16* A, int lda, int R, int H,
                                               int tid, int nthreads) {
  const int G   = nthreads / R;        // threads per row (4 or 8; divides 32)
  const int r   = tid / G;
  const int sub = tid % G;

  float s = 0.f, s2 = 0.f;
  for (int c = sub; c < H; c += G) {
    const float v = C[r * ldc + c] + bias[c];
    s += v; s2 += v * v;
  }
#pragma unroll
  for (int off = 4; off > 0; off >>= 1) {          // G <= 8
    if (off < G) {
      s  += __shfl_xor(s,  off, 32);
      s2 += __shfl_xor(s2, off, 32);
    }
  }
  s  += __shfl_xor(s,  G >> 1, 32);                // top step (off == G/2)
  s2 += __shfl_xor(s2, G >> 1, 32);

  const float mu  = s / (float)H;
  const float var = s2 / (float)H - mu * mu;
  const float inv = rsqrtf(var + 1e-5f);
  for (int c = sub; c < H; c += G) {
    const float v = (C[r * ldc + c] + bias[c] - mu) * inv * g[c] + be[c];
    A[r * lda + c] = (_Float16)gelu_f(v);
  }
  __syncthreads();
  for (int c = tid; c < H; c += nthreads) {
    float mx = -3.402823466e38f;
    for (int rr = 0; rr < R; ++rr) mx = fmaxf(mx, (float)A[rr * lda + c]);
    const _Float16 h = (_Float16)mx;
    for (int rr = 0; rr < R; ++rr) A[rr * lda + H + c] = h;
  }
  __syncthreads();
}

// -------------------- weight prep: fp32 [cin][cout] -> f16 WT [cout][cinPad] --------------------

__global__ void prep_wt_kernel(const float* __restrict__ W, _Float16* __restrict__ WT,
                               int cin, int cout, int cinPad) {
  const int i = blockIdx.x * blockDim.x + threadIdx.x;
  if (i >= cout * cinPad) return;
  const int n = i / cinPad, k = i % cinPad;
  WT[n * cinPad + k] = (_Float16)((k < cin) ? W[k * cout + n] : 0.0f);
}

// -------------------- level 1 encoder: one block (4 waves) per patch --------------------

__global__ __launch_bounds__(128) void enc1_kernel(
    const float* __restrict__ masks, const float* __restrict__ coords,
    const float* __restrict__ centers1, const int* __restrict__ knn1,
    const _Float16* __restrict__ W0T, const float* __restrict__ b0,
    const float* __restrict__ g0, const float* __restrict__ be0,
    const _Float16* __restrict__ W1T, const float* __restrict__ b1,
    const float* __restrict__ g1, const float* __restrict__ be1,
    const _Float16* __restrict__ WoT, const float* __restrict__ bo,
    float* __restrict__ x1out) {
  __shared__ _Float16 Asm[32 * 256];   // activation staging (f16, WMMA-A feed)
  __shared__ float    Csm[32 * 128];   // f32 matmul output

  const int p = blockIdx.x;            // 0 .. B*M1-1
  const int b = p / M1_;
  const int tid = threadIdx.x, lane = tid & 31, wave = tid >> 5;

  // ---- build input patch [32 x 4], K-padded to 32 ----
  for (int i = tid; i < 32 * 32; i += 128) Asm[i] = (_Float16)0.0f;
  __syncthreads();
  {
    const int r = tid >> 2, c = tid & 3;           // 128 threads = 32x4 exactly
    const int idx = knn1[(long)p * K1_ + r];
    float v;
    if (c < 3)
      v = (coords[((long)b * N_ + idx) * 3 + c] - centers1[(long)p * 3 + c]) * 10.0f; // /0.1
    else
      v = masks[(long)b * N_ + idx];
    Asm[r * 32 + c] = (_Float16)v;
  }
  __syncthreads();

  // ---- block 0: [32x32(pad 4)] @ [32x64] ----
  mm_lds(Asm, 32, W0T, 32, Csm, 64, 32, 64, 32, wave, lane);
  __syncthreads();
  ln_gelu_concat(Csm, 64, b0, g0, be0, Asm, 128, 32, 64, tid, 128);

  // ---- block 1: [32x128] @ [128x128] ----
  mm_lds(Asm, 128, W1T, 128, Csm, 128, 32, 128, 128, wave, lane);
  __syncthreads();
  ln_gelu_concat(Csm, 128, b1, g1, be1, Asm, 256, 32, 128, tid, 128);

  // ---- out proj: [32x256] @ [256x128], then max over K rows ----
  mm_lds(Asm, 256, WoT, 256, Csm, 128, 32, 128, 256, wave, lane);
  __syncthreads();
  for (int c = tid; c < 128; c += 128) {
    float mx = -3.402823466e38f;
    for (int r = 0; r < 32; ++r) mx = fmaxf(mx, Csm[r * 128 + c]);
    x1out[(long)p * 128 + c] = mx + bo[c];
  }
}

// -------------------- level 2 encoder --------------------

__global__ __launch_bounds__(128) void enc2_kernel(
    const float* __restrict__ centers1, const float* __restrict__ x1,
    const float* __restrict__ centers2, const int* __restrict__ knn2,
    const _Float16* __restrict__ W0T, const float* __restrict__ b0,
    const float* __restrict__ g0, const float* __restrict__ be0,
    const _Float16* __restrict__ W1T, const float* __restrict__ b1,
    const float* __restrict__ g1, const float* __restrict__ be1,
    const _Float16* __restrict__ WoT, const float* __restrict__ bo,
    float* __restrict__ x2out) {
  __shared__ _Float16 Asm[16 * 512];
  __shared__ float    Csm[16 * 256];

  const int p = blockIdx.x;            // 0 .. B*M2-1
  const int b = p / M2_;
  const int tid = threadIdx.x, lane = tid & 31, wave = tid >> 5;

  // ---- build input patch [16 x 131], K-padded to 160 ----
  for (int i = tid; i < 16 * 160; i += 128) Asm[i] = (_Float16)0.0f;
  __syncthreads();
  for (int e = tid; e < 16 * 131; e += 128) {
    const int r = e / 131, c = e % 131;
    const int idx = knn2[(long)p * K2_ + r];
    float v;
    if (c < 3)
      v = (centers1[((long)b * M1_ + idx) * 3 + c] - centers2[(long)p * 3 + c]) * 2.5f; // /0.4
    else
      v = x1[((long)b * M1_ + idx) * 128 + (c - 3)];
    Asm[r * 160 + c] = (_Float16)v;
  }
  __syncthreads();

  // ---- block 0: [16x160(pad 131)] @ [160x128] ----
  mm_lds(Asm, 160, W0T, 160, Csm, 128, 16, 128, 160, wave, lane);
  __syncthreads();
  ln_gelu_concat(Csm, 128, b0, g0, be0, Asm, 256, 16, 128, tid, 128);

  // ---- block 1: [16x256] @ [256x256] ----
  mm_lds(Asm, 256, W1T, 256, Csm, 256, 16, 256, 256, wave, lane);
  __syncthreads();
  ln_gelu_concat(Csm, 256, b1, g1, be1, Asm, 512, 16, 256, tid, 128);

  // ---- out proj: [16x512] @ [512x256], then max over K rows ----
  mm_lds(Asm, 512, WoT, 512, Csm, 256, 16, 256, 512, wave, lane);
  __syncthreads();
  for (int c = tid; c < 256; c += 128) {
    float mx = -3.402823466e38f;
    for (int r = 0; r < 16; ++r) mx = fmaxf(mx, Csm[r * 256 + c]);
    x2out[(long)p * 256 + c] = mx + bo[c];
  }
}

// -------------------- launch --------------------

extern "C" void kernel_launch(void* const* d_in, const int* in_sizes, int n_in,
                              void* d_out, int out_size, void* d_ws, size_t ws_size,
                              hipStream_t stream) {
  (void)in_sizes; (void)n_in; (void)out_size; (void)ws_size;

  const float* masks    = (const float*)d_in[0];
  const float* coords   = (const float*)d_in[1];
  const float* centers1 = (const float*)d_in[2];
  const int*   knn1     = (const int*)d_in[3];
  const float* centers2 = (const float*)d_in[4];
  const int*   knn2     = (const int*)d_in[5];
  // enc1 params (pytree leaf order)
  const float* e1w0 = (const float*)d_in[6];
  const float* e1b0 = (const float*)d_in[7];
  const float* e1g0 = (const float*)d_in[8];
  const float* e1e0 = (const float*)d_in[9];
  const float* e1w1 = (const float*)d_in[10];
  const float* e1b1 = (const float*)d_in[11];
  const float* e1g1 = (const float*)d_in[12];
  const float* e1e1 = (const float*)d_in[13];
  const float* e1wo = (const float*)d_in[14];
  const float* e1bo = (const float*)d_in[15];
  // enc2 params
  const float* e2w0 = (const float*)d_in[16];
  const float* e2b0 = (const float*)d_in[17];
  const float* e2g0 = (const float*)d_in[18];
  const float* e2e0 = (const float*)d_in[19];
  const float* e2w1 = (const float*)d_in[20];
  const float* e2b1 = (const float*)d_in[21];
  const float* e2g1 = (const float*)d_in[22];
  const float* e2e1 = (const float*)d_in[23];
  const float* e2wo = (const float*)d_in[24];
  const float* e2bo = (const float*)d_in[25];

  // workspace layout (f16 elements): transposed/padded weights
  _Float16* ws = (_Float16*)d_ws;
  _Float16* W0T1 = ws;                       // [64][32]
  _Float16* W1T1 = W0T1 + 64 * 32;           // [128][128]
  _Float16* WoT1 = W1T1 + 128 * 128;         // [128][256]
  _Float16* W0T2 = WoT1 + 128 * 256;         // [128][160]
  _Float16* W1T2 = W0T2 + 128 * 160;         // [256][256]
  _Float16* WoT2 = W1T2 + 256 * 256;         // [256][512]

  prep_wt_kernel<<<(64 * 32 + 255) / 256, 256, 0, stream>>>(e1w0, W0T1,   4,  64,  32);
  prep_wt_kernel<<<(128 * 128 + 255) / 256, 256, 0, stream>>>(e1w1, W1T1, 128, 128, 128);
  prep_wt_kernel<<<(128 * 256 + 255) / 256, 256, 0, stream>>>(e1wo, WoT1, 256, 128, 256);
  prep_wt_kernel<<<(128 * 160 + 255) / 256, 256, 0, stream>>>(e2w0, W0T2, 131, 128, 160);
  prep_wt_kernel<<<(256 * 256 + 255) / 256, 256, 0, stream>>>(e2w1, W1T2, 256, 256, 256);
  prep_wt_kernel<<<(256 * 512 + 255) / 256, 256, 0, stream>>>(e2wo, WoT2, 512, 256, 512);

  float* x1out = (float*)d_out;                       // [B,M1,128]
  float* x2out = x1out + (long)B_ * M1_ * 128;        // [B,M2,256]

  enc1_kernel<<<B_ * M1_, 128, 0, stream>>>(masks, coords, centers1, knn1,
                                            W0T1, e1b0, e1g0, e1e0,
                                            W1T1, e1b1, e1g1, e1e1,
                                            WoT1, e1bo, x1out);

  enc2_kernel<<<B_ * M2_, 128, 0, stream>>>(centers1, x1out, centers2, knn2,
                                            W0T2, e2b0, e2g0, e2e0,
                                            W1T2, e2b1, e2g1, e2e1,
                                            WoT2, e2bo, x2out);
}
